// MinGRU_13237089206866
// MI455X (gfx1250) — matrix operand order repaired and checked by reading, not compiled, not run on hardware
//
#include <hip/hip_runtime.h>
#include <hip/hip_bf16.h>
#include <math.h>

// ---------------- problem sizes ----------------
#define BQ   4
#define LQ   4096
#define DQ   2048
#define MDIM (BQ * LQ)        // 16384 rows
#define KDIM DQ               // 2048
#define NDIM DQ               // 2048

// GEMM tiling
#define TM 128
#define TN 64
#define BK 32                 // one bf16 WMMA K-step
#define KT (KDIM / BK)        // 64

// scan chunking
#define CH 16
#define TT (LQ / CH)          // 256

typedef __attribute__((ext_vector_type(8)))  float  v8f;
typedef __attribute__((ext_vector_type(4)))  int    v4i;
typedef __attribute__((ext_vector_type(16))) __bf16 bf16x16;
typedef __attribute__((ext_vector_type(8)))  __bf16 bf16x8;
typedef __attribute__((ext_vector_type(4)))  __bf16 bf16x4;

#if defined(__AMDGCN__) && __has_builtin(__builtin_amdgcn_global_load_async_to_lds_b128)
#define USE_ASYNC_LDS 1
#else
#define USE_ASYNC_LDS 0
#endif

#if defined(__AMDGCN__)
typedef v4i __attribute__((address_space(1)))* gptr4;
typedef v4i __attribute__((address_space(3)))* lptr4;
#endif

__device__ __forceinline__ void async_copy16(const void* g, void* l) {
#if USE_ASYNC_LDS
    // AS(1) = global (full 64-bit address), AS(3) = LDS (offset lives in low 32 bits
    // of the generic pointer; shared aperture occupies the high 32 bits).
    __builtin_amdgcn_global_load_async_to_lds_b128(
        (gptr4)(uintptr_t)g,
        (lptr4)(uint32_t)(uintptr_t)l,
        0, 0);
#else
    *(float4*)l = *(const float4*)g;   // synchronous fallback: vmem load + ds store
#endif
}

template <int N>
__device__ __forceinline__ void wait_async() {
#if USE_ASYNC_LDS
#if __has_builtin(__builtin_amdgcn_s_wait_asynccnt)
    __builtin_amdgcn_s_wait_asynccnt(N);
#else
    asm volatile("s_wait_asynccnt %0" ::"i"(N) : "memory");
#endif
#endif
}

// -------------------------------------------------------------------------
// fp32 -> bf16 (RNE via hardware convert), 4 elements per thread
// -------------------------------------------------------------------------
__global__ void cvt_f32_bf16(const float* __restrict__ in, __bf16* __restrict__ out, int n4) {
    int i = blockIdx.x * blockDim.x + threadIdx.x;
    if (i >= n4) return;
    float4 f = ((const float4*)in)[i];
    bf16x4 o;
    o.x = (__bf16)f.x; o.y = (__bf16)f.y; o.z = (__bf16)f.z; o.w = (__bf16)f.w;
    ((bf16x4*)out)[i] = o;
}

// -------------------------------------------------------------------------
// Dual GEMM (k = x Wz^T, n = x Wn^T) + fused log-space epilogue.
// Block: 256 threads (8 waves), tile 128(M) x 64(N), K pipelined in 32-chunks
// with async global->LDS double buffering. Each wave: 2x2 WMMA tiles x 2 mats.
// -------------------------------------------------------------------------
__global__ void __launch_bounds__(256)
dual_gemm_log_epilogue(const __bf16* __restrict__ xb,
                       const __bf16* __restrict__ wzb,
                       const __bf16* __restrict__ wnb,
                       const float* __restrict__ bz,
                       const float* __restrict__ bn,
                       float* __restrict__ lc,
                       float* __restrict__ lv) {
    __shared__ __align__(128) __bf16 smem[2][TM * BK + 2 * TN * BK];

    const int t     = threadIdx.x;
    const int mBase = blockIdx.y * TM;
    const int nBase = blockIdx.x * TN;
    const int lane  = t & 31;
    const int wave  = t >> 5;
    const int wm    = wave & 3;   // 0..3 -> 32-row slice
    const int wnn   = wave >> 2;  // 0..1 -> 32-col slice
    const int khalf = lane >> 4;  // K-half selector per WMMA layout
    const int lrow  = lane & 15;

    v8f accz[2][2], accn[2][2];
    const v8f zero = {0.f, 0.f, 0.f, 0.f, 0.f, 0.f, 0.f, 0.f};
#pragma unroll
    for (int mt = 0; mt < 2; ++mt)
#pragma unroll
        for (int nt = 0; nt < 2; ++nt) { accz[mt][nt] = zero; accn[mt][nt] = zero; }

    // issue 4 async b128 loads per thread per buffer fill (A: 2, Bz: 1, Bn: 1)
    auto fill = [&](int buf, int kk) {
        __bf16* As  = &smem[buf][0];
        __bf16* Bzs = As + TM * BK;
        __bf16* Bns = Bzs + TN * BK;
        const int row  = t >> 2;      // 0..63
        const int part = t & 3;       // 16B sub-chunk of a 64B row
        async_copy16(xb + (size_t)(mBase + row) * KDIM + kk + part * 8,
                     As + row * BK + part * 8);
        const int row2 = row + 64;    // rows 64..127 of A tile
        async_copy16(xb + (size_t)(mBase + row2) * KDIM + kk + part * 8,
                     As + row2 * BK + part * 8);
        async_copy16(wzb + (size_t)(nBase + row) * KDIM + kk + part * 8,
                     Bzs + row * BK + part * 8);
        async_copy16(wnb + (size_t)(nBase + row) * KDIM + kk + part * 8,
                     Bns + row * BK + part * 8);
    };

    fill(0, 0);

    for (int kt = 0; kt < KT; ++kt) {
        const int cur = kt & 1;
        if (kt + 1 < KT) {
            fill(cur ^ 1, (kt + 1) * BK);
            wait_async<4>();          // the 4 newer loads may stay in flight
        } else {
            wait_async<0>();
        }
        __syncthreads();

        const __bf16* As  = &smem[cur][0];
        const __bf16* Bzs = As + TM * BK;
        const __bf16* Bns = Bzs + TN * BK;

        // A fragments (16-bit A layout: lane holds row lane%16; K split by lane/16)
        bf16x16 afrag[2];
#pragma unroll
        for (int mt = 0; mt < 2; ++mt) {
            const __bf16* p = As + (wm * 32 + mt * 16 + lrow) * BK + khalf * 16;
            union { bf16x16 v; bf16x8 h[2]; } u;
            u.h[0] = *(const bf16x8*)(p);
            u.h[1] = *(const bf16x8*)(p + 8);
            afrag[mt] = u.v;
        }
        // B fragments (lane holds column lane%16, contiguous K16 block by lane/16)
        bf16x16 bzf[2], bnf[2];
#pragma unroll
        for (int nt = 0; nt < 2; ++nt) {
            const int off = (wnn * 32 + nt * 16 + lrow) * BK + khalf * 16;
            bzf[nt] = *(const bf16x16*)(Bzs + off);
            bnf[nt] = *(const bf16x16*)(Bns + off);
        }
#pragma unroll
        for (int mt = 0; mt < 2; ++mt)
#pragma unroll
            for (int nt = 0; nt < 2; ++nt) {
                accz[mt][nt] = __builtin_amdgcn_wmma_f32_16x16x32_bf16(
                    false, afrag[mt], false, bzf[nt], (short)0, accz[mt][nt], false, false);
                accn[mt][nt] = __builtin_amdgcn_wmma_f32_16x16x32_bf16(
                    false, afrag[mt], false, bnf[nt], (short)0, accn[mt][nt], false, false);
            }
        __syncthreads();
    }

    // fused epilogue: lc = -softplus(k) ; lv = -softplus(-k) + log_g(n)
#pragma unroll
    for (int mt = 0; mt < 2; ++mt)
#pragma unroll
        for (int nt = 0; nt < 2; ++nt) {
            const int col  = nBase + wnn * 32 + nt * 16 + lrow;
            const float bzv = bz[col];
            const float bnv = bn[col];
            const int row0 = mBase + wm * 32 + mt * 16 + khalf * 8;
#pragma unroll
            for (int r = 0; r < 8; ++r) {
                const size_t idx = (size_t)(row0 + r) * NDIM + col;
                const float kv = accz[mt][nt][r] + bzv;
                const float nv = accn[mt][nt][r] + bnv;
                const float sp_k  = log1pf(expf(-fabsf(kv))) + fmaxf(kv, 0.f); // softplus(k)
                const float sp_mk = sp_k - kv;                                  // softplus(-k)
                const float lg = (nv >= 0.f)
                                     ? logf(nv + 0.5f)
                                     : -(log1pf(expf(-fabsf(nv))) + fmaxf(-nv, 0.f));
                lc[idx] = -sp_k;
                lv[idx] = -sp_mk + lg;   // log_z + log_h_
            }
        }
}

// -------------------------------------------------------------------------
// Chunked log-space scan: G_t = logaddexp(G_{t-1} + lc_t, lv_t), G_0 = log(0.5)
// -------------------------------------------------------------------------
__device__ __forceinline__ float logaddexp_(float a, float b) {
    const float m = fmaxf(a, b);
    return m + log1pf(expf(fminf(a, b) - m));   // expf(-inf) == 0 handles H=-inf
}

__global__ void scan_pass1(const float* __restrict__ lc, const float* __restrict__ lv,
                           float* __restrict__ Cc, float* __restrict__ Vv) {
    const int gid  = blockIdx.x * blockDim.x + threadIdx.x;  // B*CH*D threads
    const int d    = gid & (DQ - 1);
    const int rest = gid >> 11;
    const int ch   = rest & (CH - 1);
    const int b    = rest >> 4;
    size_t idx = (size_t)(b * LQ + ch * TT) * DQ + d;
    float Cs = 0.f, H = -INFINITY;
    for (int i = 0; i < TT; ++i, idx += DQ) {
        const float c = lc[idx];
        const float v = lv[idx];
        Cs += c;
        H = logaddexp_(H + c, v);
    }
    Cc[gid] = Cs;
    Vv[gid] = H;
}

__global__ void scan_pass2(const float* __restrict__ Cc, const float* __restrict__ Vv,
                           float* __restrict__ Pp) {
    const int gid = blockIdx.x * blockDim.x + threadIdx.x;   // B*D threads
    const int d = gid & (DQ - 1);
    const int b = gid >> 11;
    float G = -0.69314718055994531f;                          // log(0.5) = log_g(0)
    for (int ch = 0; ch < CH; ++ch) {
        const int idx = (b * CH + ch) * DQ + d;
        Pp[idx] = G;                                          // state entering chunk
        G = logaddexp_(G + Cc[idx], Vv[idx]);
    }
}

__global__ void scan_pass3(const float* __restrict__ lc, const float* __restrict__ lv,
                           const float* __restrict__ Pp, float* __restrict__ out) {
    const int gid  = blockIdx.x * blockDim.x + threadIdx.x;
    const int d    = gid & (DQ - 1);
    const int rest = gid >> 11;
    const int ch   = rest & (CH - 1);
    const int b    = rest >> 4;
    size_t idx = (size_t)(b * LQ + ch * TT) * DQ + d;
    float G = Pp[gid];
    for (int i = 0; i < TT; ++i, idx += DQ) {
        G = logaddexp_(G + lc[idx], lv[idx]);   // lv may alias out: read-before-write
        out[idx] = expf(G);
    }
}

// -------------------------------------------------------------------------
extern "C" void kernel_launch(void* const* d_in, const int* in_sizes, int n_in,
                              void* d_out, int out_size, void* d_ws, size_t ws_size,
                              hipStream_t stream) {
    const float* x  = (const float*)d_in[0];
    const float* Wz = (const float*)d_in[1];
    const float* bz = (const float*)d_in[2];
    const float* Wn = (const float*)d_in[3];
    const float* bn = (const float*)d_in[4];
    float* out = (float*)d_out;

    const size_t XB  = (size_t)MDIM * KDIM * 2;   // x in bf16: 64 MiB
    const size_t WB  = (size_t)NDIM * KDIM * 2;   // W in bf16: 8 MiB each
    const size_t LCB = (size_t)MDIM * NDIM * 4;   // lc fp32: 128 MiB
    char* ws = (char*)d_ws;
    __bf16* xb  = (__bf16*)(ws);
    __bf16* wzb = (__bf16*)(ws + XB);
    __bf16* wnb = (__bf16*)(ws + XB + WB);
    float*  lc  = (float*)(ws + XB + 2 * WB);
    float*  Cc  = (float*)(ws + XB + 2 * WB + LCB);
    float*  Vv  = Cc + BQ * DQ * CH;
    float*  Pp  = Vv + BQ * DQ * CH;
    float*  lv  = out;                             // alias: saves 128 MiB, pass3 reads first

    // 1) fp32 -> bf16
    cvt_f32_bf16<<<(MDIM * KDIM / 4) / 256, 256, 0, stream>>>(x, xb, MDIM * KDIM / 4);
    cvt_f32_bf16<<<(NDIM * KDIM / 4) / 256, 256, 0, stream>>>(Wz, wzb, NDIM * KDIM / 4);
    cvt_f32_bf16<<<(NDIM * KDIM / 4) / 256, 256, 0, stream>>>(Wn, wnb, NDIM * KDIM / 4);

    // 2) dual GEMM + log-space epilogue
    dual_gemm_log_epilogue<<<dim3(NDIM / TN, MDIM / TM), 256, 0, stream>>>(
        xb, wzb, wnb, bz, bn, lc, lv);

    // 3) chunked scan
    scan_pass1<<<(BQ * CH * DQ) / 256, 256, 0, stream>>>(lc, lv, Cc, Vv);
    scan_pass2<<<(BQ * DQ) / 256, 256, 0, stream>>>(Cc, Vv, Pp);
    scan_pass3<<<(BQ * CH * DQ) / 256, 256, 0, stream>>>(lc, lv, Pp, out);
}
